// Contextualizer_11424613007992
// MI455X (gfx1250) — matrix operand order repaired and checked by reading, not compiled, not run on hardware
//
#include <hip/hip_runtime.h>
#include <hip/hip_bf16.h>

// ---------------------------------------------------------------------------
// Fused "contextualizer": out = (tril(P) * cosim(x0)) @ x0 * x1
//   B=8, N=2048, D=1024 (x = [B,N,2D])
// Compute-bound -> f16 WMMA w/ f32 accumulate; flash-style fusion (the
// [B,N,N] attention matrix never touches global memory).
// v6: hard sched_barrier fences (VMEM+WMMA pinned, ALU/DS free) between the
//     k+1 B-load and the k wmma pair so the 2-deep double buffer survives
//     scheduling + register allocation (wait becomes loadcnt<=2, not 0).
// Workspace: xn (f16, 32MB) + xn^T (f16, 32MB) + norms (64KB).
// ---------------------------------------------------------------------------

typedef __attribute__((ext_vector_type(16))) _Float16 v16h;
typedef __attribute__((ext_vector_type(8)))  float    v8f;

static constexpr int kB  = 8;
static constexpr int kN  = 2048;
static constexpr int kD  = 1024;
static constexpr int kD2 = 2048;

static constexpr int kSaStride = 136;    // 128 + 8 halves pad (16B aligned)
static constexpr int kAStride  = 1032;   // 1024 + 8 halves pad (16B aligned)

// sched_barrier: set bits MAY cross.  Allow ALU/VALU/SALU/DS (0x387);
// pin VMEM (0x10|0x20|0x40) and MFMA/WMMA (0x8).
#define SB_FENCE() __builtin_amdgcn_sched_barrier(0x387)

// ---------------------------------------------------------------------------
// Kernel 1a: per-row L2 norm of x0 (sqrt(sum(x0^2) + eps))
// ---------------------------------------------------------------------------
__global__ __launch_bounds__(256) void row_norms_kernel(
    const float* __restrict__ x, float* __restrict__ norms)
{
    __shared__ float red[256];
    const int row = blockIdx.x;                    // b*N + n
    const float* xr = x + (size_t)row * kD2;       // x0 = first D columns
    float s = 0.f;
    for (int i = threadIdx.x; i < kD; i += 256) {
        float v = xr[i];
        s += v * v;
    }
    red[threadIdx.x] = s;
    __syncthreads();
    for (int off = 128; off > 0; off >>= 1) {
        if (threadIdx.x < off) red[threadIdx.x] += red[threadIdx.x + off];
        __syncthreads();
    }
    if (threadIdx.x == 0) norms[row] = sqrtf(red[0] + 1e-8f);
}

// ---------------------------------------------------------------------------
// Kernel 1b: xn = (x0 / norm) as f16, stored row-major AND transposed.
// ---------------------------------------------------------------------------
__global__ __launch_bounds__(256) void make_xn_kernel(
    const float* __restrict__ x, const float* __restrict__ norms,
    _Float16* __restrict__ xnh, _Float16* __restrict__ xnT)
{
    __shared__ _Float16 tile[64][65];              // +1 pad vs bank conflicts
    const int blk = blockIdx.x;
    const int dt  = blk & 15;                      // D/64 = 16
    const int nt  = (blk >> 4) & 31;               // N/64 = 32
    const int b   = blk >> 9;
    const int n0  = nt << 6, d0 = dt << 6;
    const float* xb = x + (size_t)b * kN * kD2;

    #pragma unroll
    for (int k = 0; k < 16; ++k) {
        int idx = threadIdx.x + (k << 8);
        int rn = idx >> 6, cd = idx & 63;
        float rinv = 1.0f / norms[b * kN + n0 + rn];
        _Float16 h = (_Float16)(xb[(size_t)(n0 + rn) * kD2 + d0 + cd] * rinv);
        tile[rn][cd] = h;
        xnh[((size_t)(b * kN + n0 + rn)) * kD + d0 + cd] = h;
    }
    __syncthreads();
    #pragma unroll
    for (int k = 0; k < 16; ++k) {
        int idx = threadIdx.x + (k << 8);
        int rd = idx >> 6, cn = idx & 63;
        xnT[((size_t)(b * kD + d0 + rd)) * kN + n0 + cn] = tile[cn][rd];
    }
}

// ---------------------------------------------------------------------------
// gfx1250 async copy helpers (ASYNCcnt-tracked, no VGPR round-trip)
// ---------------------------------------------------------------------------
static __device__ __forceinline__ void async_load_b128(uint32_t lds_off,
                                                       const void* gptr)
{
    asm volatile("global_load_async_to_lds_b128 %0, %1, off"
                 :: "v"(lds_off), "v"(gptr) : "memory");
}
static __device__ __forceinline__ void wait_async0()
{
    asm volatile("s_wait_asynccnt 0x0" ::: "memory");
}

// ---------------------------------------------------------------------------
// Kernel 2: fused causal GEMM-mask-GEMM.
// Block = 256 threads = 8 waves; handles (batch b, 32-row tile i0).
// A-tile (32x1024 f16) lives in LDS (async-staged once, reused every j-tile).
// Loop over 128-wide m-tiles (causal):
//   GEMM1: wave w owns m-slice [j0+16w, +16): two 16x16 S tiles via 64 wmma;
//          B operand double-buffered in registers, feeds both subtiles.
//   mask : tril * P[n,m] * norm[m] on register accumulators -> f16 Sa in LDS.
//   GEMM2: wave w owns d-slice [128w, +128): ctx(32x128) += Sa @ xn_j via
//          64 wmma; B operand double-buffered, feeds both subtiles.
// ---------------------------------------------------------------------------
__global__ __launch_bounds__(256) void ctx_main_kernel(
    const float* __restrict__ x, const float* __restrict__ P,
    const _Float16* __restrict__ xnh, const _Float16* __restrict__ xnT,
    const float* __restrict__ norms, float* __restrict__ out)
{
    __shared__ __align__(16) _Float16 Alds[32 * kAStride];  // ~64.5 KB
    __shared__ __align__(16) _Float16 Sa[32 * kSaStride];   // ~8.5 KB

    const int tid   = threadIdx.x;
    const int wave  = tid >> 5;
    const int lane  = tid & 31;
    const int lhalf = lane >> 4;        // 0: lanes 0-15, 1: lanes 16-31
    const int lmod  = lane & 15;

    const int b  = blockIdx.x >> 6;     // N/32 = 64 row tiles per batch
    const int i0 = (blockIdx.x & 63) << 5;

    const int mw = wave << 4;           // wave's m-slice inside the 128 tile
    const int dw = wave << 7;           // wave's d-slice (GEMM2 / output)

    const _Float16* xn_b  = xnh + (size_t)b * kN * kD;
    const _Float16* xnT_b = xnT + (size_t)b * kD * kN;

    // ---- stage A-tile: xn rows i0..i0+31 -> LDS (async, 16B granules).
    {
        for (int c = tid; c < 32 * 64; c += 256) {        // 2048 granules
            const int row = c >> 6, ch = c & 63;
            const _Float16* g = xn_b + (size_t)(i0 + row) * kD + (ch << 4);
            const uint32_t loff =
                (uint32_t)(size_t)&Alds[row * kAStride + (ch << 4)];
            async_load_b128(loff, g);
        }
        wait_async0();
        __syncthreads();
    }

    v8f acc[2][8];
    #pragma unroll
    for (int t = 0; t < 2; ++t)
        #pragma unroll
        for (int c = 0; c < 8; ++c) acc[t][c] = (v8f){};

    // A-operand row pointers in LDS (A layout: lane<16 holds M=lane,
    // K {0..7,16..23}; lane>=16 holds M=lane-16, K {8..15,24..31}).
    const _Float16* arow_lo = &Alds[lmod * kAStride + (lhalf ? 8 : 0)];
    const _Float16* arow_hi = &Alds[(16 + lmod) * kAStride + (lhalf ? 8 : 0)];

    union U { v16h v; uint4 q[2]; };

    // causal: only m-tiles with m_base <= i0+31 contribute
    const int jtiles = (i0 + 32 + 127) >> 7;
    for (int jt = 0; jt < jtiles; ++jt) {
        const int j0 = jt << 7;

        // prefetch next tile's B streams (global_prefetch_b8)
        if (jt + 1 < jtiles) {
            __builtin_prefetch(
                xn_b + (size_t)(j0 + 128 + mw + lmod) * kD, 0, 1);
            __builtin_prefetch(
                xnT_b + (size_t)(dw + (lhalf << 3) + lmod) * kN + j0 + 128, 0, 1);
        }

        // ---- GEMM1: two full-K 16x16 S tiles for this wave's m-slice.
        // B operand column Ncol = row m of xn; lanes 0-15 hold K {0..7,16..23},
        // lanes 16-31 hold K {8..15,24..31}.  2-deep double buffer on B; a hard
        // fence keeps load(k+1) ABOVE the wmmas of step k.
        v8f s0 = (v8f){}, s1 = (v8f){};
        {
            const _Float16* brow =
                xn_b + (size_t)(j0 + mw + lmod) * kD + (lhalf ? 8 : 0);
            U ub[2];
            ub[0].q[0] = *(const uint4*)(brow);
            ub[0].q[1] = *(const uint4*)(brow + 16);
            #pragma unroll
            for (int k = 0; k < 32; ++k) {
                if (k + 1 < 32) {
                    ub[(k + 1) & 1].q[0] = *(const uint4*)(brow + 32 * (k + 1));
                    ub[(k + 1) & 1].q[1] = *(const uint4*)(brow + 32 * (k + 1) + 16);
                }
                SB_FENCE();           // load(k+1) stays above wmma(k)
                U ua0, ua1;
                ua0.q[0] = *(const uint4*)(arow_lo + 32 * k);
                ua0.q[1] = *(const uint4*)(arow_lo + 32 * k + 16);
                ua1.q[0] = *(const uint4*)(arow_hi + 32 * k);
                ua1.q[1] = *(const uint4*)(arow_hi + 32 * k + 16);
                s0 = __builtin_amdgcn_wmma_f32_16x16x32_f16(
                    false, ua0.v, false, ub[k & 1].v, (short)0, s0, false, false);
                s1 = __builtin_amdgcn_wmma_f32_16x16x32_f16(
                    false, ua1.v, false, ub[k & 1].v, (short)0, s1, false, false);
            }
        }

        __syncthreads();   // previous iteration's Sa consumers are done

        // ---- mask: tril * P[n,m] * norm[m] -> f16 Sa (32 x 128).
        // C layout: element r of lane = (M = r + 8*lhalf, Ncol = lmod).
        {
            const int m = j0 + mw + lmod;
            const float nm = norms[b * kN + m];
            #pragma unroll
            for (int r = 0; r < 8; ++r) {
                const int rr = r + (lhalf << 3);
                const int n0r = i0 + rr;          // subtile 0 row
                const int n1r = i0 + 16 + rr;     // subtile 1 row
                float v0 = 0.f, v1 = 0.f;
                if (m <= n0r) v0 = s0[r] * P[(size_t)n0r * kN + m] * nm;
                if (m <= n1r) v1 = s1[r] * P[(size_t)n1r * kN + m] * nm;
                Sa[rr * kSaStride + mw + lmod]        = (_Float16)v0;
                Sa[(16 + rr) * kSaStride + mw + lmod] = (_Float16)v1;
            }
        }
        __syncthreads();

        // ---- GEMM2: ctx(32 x 128) += Sa(32x128) @ xn_j(128 x 128d).
        // A operands hoisted from LDS; B double-buffered from xn^T with the
        // same fence-enforced skew.
        U ua[2][4];
        #pragma unroll
        for (int t = 0; t < 2; ++t)
            #pragma unroll
            for (int k = 0; k < 4; ++k) {
                const _Float16* sp =
                    &Sa[((t << 4) + lmod) * kSaStride + 32 * k + (lhalf ? 8 : 0)];
                ua[t][k].q[0] = *(const uint4*)(sp);
                ua[t][k].q[1] = *(const uint4*)(sp + 16);
            }
        {
            const _Float16* bbase =
                xnT_b + (size_t)(dw + lmod) * kN + j0 + (lhalf ? 8 : 0);
            U ub[2];
            ub[0].q[0] = *(const uint4*)(bbase);
            ub[0].q[1] = *(const uint4*)(bbase + 16);
            #pragma unroll
            for (int idx = 0; idx < 32; ++idx) {
                const int c = idx >> 2, k = idx & 3;
                if (idx + 1 < 32) {
                    const int c1 = (idx + 1) >> 2, k1 = (idx + 1) & 3;
                    const _Float16* br =
                        bbase + (size_t)(c1 << 4) * kN + 32 * k1;
                    ub[(idx + 1) & 1].q[0] = *(const uint4*)(br);
                    ub[(idx + 1) & 1].q[1] = *(const uint4*)(br + 16);
                }
                SB_FENCE();           // load(idx+1) stays above wmma(idx)
                acc[0][c] = __builtin_amdgcn_wmma_f32_16x16x32_f16(
                    false, ua[0][k].v, false, ub[idx & 1].v, (short)0,
                    acc[0][c], false, false);
                acc[1][c] = __builtin_amdgcn_wmma_f32_16x16x32_f16(
                    false, ua[1][k].v, false, ub[idx & 1].v, (short)0,
                    acc[1][c], false, false);
            }
        }
    }

    // ---- epilogue: out = ctx * x1   (C layout: M = r + 8*lhalf, Ncol = lmod)
    #pragma unroll
    for (int t = 0; t < 2; ++t)
        #pragma unroll
        for (int c = 0; c < 8; ++c)
            #pragma unroll
            for (int r = 0; r < 8; ++r) {
                const int n = i0 + (t << 4) + r + (lhalf << 3);
                const int d = dw + (c << 4) + lmod;
                const size_t rowbase = (size_t)(b * kN + n);
                const float x1 = x[rowbase * kD2 + kD + d];
                out[rowbase * kD + d] = acc[t][c][r] * x1;
            }
}

// ---------------------------------------------------------------------------
extern "C" void kernel_launch(void* const* d_in, const int* in_sizes, int n_in,
                              void* d_out, int out_size, void* d_ws, size_t ws_size,
                              hipStream_t stream) {
    const float* x = (const float*)d_in[0];        // [B,N,2D] f32
    const float* P = (const float*)d_in[1];        // [N,N]    f32
    float* out = (float*)d_out;                    // [B,N,D]  f32

    // workspace layout: xn f16 (32MB) | xn^T f16 (32MB) | norms f32 (64KB)
    char* ws = (char*)d_ws;
    const size_t xn_bytes = (size_t)kB * kN * kD * sizeof(_Float16);
    _Float16* xnh = (_Float16*)(ws);
    _Float16* xnT = (_Float16*)(ws + xn_bytes);
    float* norms  = (float*)(ws + 2 * xn_bytes);

    row_norms_kernel<<<kB * kN, 256, 0, stream>>>(x, norms);
    make_xn_kernel<<<kB * (kN / 64) * (kD / 64), 256, 0, stream>>>(x, norms, xnh, xnT);
    ctx_main_kernel<<<kB * (kN / 32), 256, 0, stream>>>(x, P, xnh, xnT, norms, out);
}
